// ChannelPriorUniScale_24180665876833
// MI455X (gfx1250) — matrix operand order repaired and checked by reading, not compiled, not run on hardware
//
#include <hip/hip_runtime.h>
#include <hip/hip_bf16.h>
#include <math.h>
#include <stdint.h>

typedef __attribute__((ext_vector_type(16))) _Float16 v16h;
typedef __attribute__((ext_vector_type(8)))  _Float16 v8h;
typedef __attribute__((ext_vector_type(8)))  float    v8f;

// ---- TDM availability probes (device pass only; host pass falls back) ----
#if defined(__has_builtin)
# if __has_builtin(__builtin_amdgcn_tensor_load_to_lds)
#  define HAVE_TDM 1
# endif
#endif
#ifndef HAVE_TDM
# define HAVE_TDM 0
#endif
#if HAVE_TDM
# if __has_include(<hip/amd_detail/amd_gfx1250_TDM.h>)
#  define TDM_SIX_ARG 1
# else
#  define TDM_SIX_ARG 0
# endif
typedef unsigned int v4u __attribute__((ext_vector_type(4)));
typedef int          v8i __attribute__((ext_vector_type(8)));
typedef int          v4i __attribute__((ext_vector_type(4)));
#endif

#define BB 32
#define SS 6
#define HH 64
#define WW 64
#define EMB 32
#define CP 40          // padded channel pitch in packed xh buffer (h:0..31, x:32..36, pad:37..39)
#define PW 72          // padded spatial extent (64 + 2*4 halo)
#define NCHUNK 29      // 25 h-tap chunks (K=32 each) + 4 x chunks (125 K padded to 128)
#define C_LOG2PI 1.8378770664093453f

// ---- workspace layout (bytes) ----
#define SZ_OUT1   ((size_t)BB*32*HH*WW*4)            // 16,777,216
#define SZ_Z1E    ((size_t)BB*4*HH*WW*4)             //  2,097,152
#define SZ_FRAGB  ((size_t)NCHUNK*128*32*2)          //    237,568
#define SZ_CBUF   ((size_t)BB*HH*WW*EMB*4)           // 16,777,216
#define SZ_XH     ((size_t)BB*PW*PW*CP*2)            // 13,271,040
#define OFF_OUT1  ((size_t)0)
#define OFF_Z1E   (OFF_OUT1 + SZ_OUT1)
#define OFF_FRAGB (OFF_Z1E + SZ_Z1E)
#define OFF_CBUF  (OFF_FRAGB + SZ_FRAGB)
#define OFF_XH0   (OFF_CBUF + SZ_CBUF)
#define OFF_XH1   (OFF_XH0 + SZ_XH)

#define PATCH_HALVES (9*PW*CP)                       // 25,920
#define SMEM_PATCH_BYTES (PATCH_HALVES*2)            // 51,840
#define SMEM_G_BYTES (64*128*4)                      // 32,768 gate buffer
#define SMEM_BYTES (SMEM_PATCH_BYTES + SMEM_G_BYTES + 256)  // + x-gather offset table

__device__ __forceinline__ float sigm(float x) { return 1.0f / (1.0f + __expf(-x)); }

// ---------------- z1 embedding: conv1 (6->32, 5x5, pad2) + ReLU ----------------
__global__ void conv_embed1(const float* __restrict__ z1, const float* __restrict__ W1,
                            const float* __restrict__ b1, float* __restrict__ out1) {
  int idx = blockIdx.x * 256 + threadIdx.x;
  if (idx >= BB * HH * WW) return;
  int x = idx % WW, y = (idx / WW) % HH, b = idx / (WW * HH);
  float acc[32];
#pragma unroll
  for (int oc = 0; oc < 32; ++oc) acc[oc] = b1[oc];
  for (int ic = 0; ic < 6; ++ic)
    for (int ky = 0; ky < 5; ++ky) {
      int iy = y + ky - 2; if (iy < 0 || iy >= HH) continue;
      for (int kx = 0; kx < 5; ++kx) {
        int ix = x + kx - 2; if (ix < 0 || ix >= WW) continue;
        float v = z1[(((size_t)b*6 + ic)*HH + iy)*WW + ix];
#pragma unroll
        for (int oc = 0; oc < 32; ++oc)
          acc[oc] += W1[((oc*6 + ic)*5 + ky)*5 + kx] * v;
      }
    }
  for (int oc = 0; oc < 32; ++oc)
    out1[(((size_t)b*32 + oc)*HH + y)*WW + x] = fmaxf(acc[oc], 0.0f);
}

// ---------------- conv2 (32->4, 5x5, pad2) ----------------
__global__ void conv_embed2(const float* __restrict__ in, const float* __restrict__ W2,
                            const float* __restrict__ b2, float* __restrict__ z1e) {
  int idx = blockIdx.x * 256 + threadIdx.x;
  if (idx >= BB * HH * WW) return;
  int x = idx % WW, y = (idx / WW) % HH, b = idx / (WW * HH);
  float acc[4] = { b2[0], b2[1], b2[2], b2[3] };
  for (int ic = 0; ic < 32; ++ic)
    for (int ky = 0; ky < 5; ++ky) {
      int iy = y + ky - 2; if (iy < 0 || iy >= HH) continue;
      for (int kx = 0; kx < 5; ++kx) {
        int ix = x + kx - 2; if (ix < 0 || ix >= WW) continue;
        float v = in[(((size_t)b*32 + ic)*HH + iy)*WW + ix];
#pragma unroll
        for (int oc = 0; oc < 4; ++oc)
          acc[oc] += W2[((oc*32 + ic)*5 + ky)*5 + kx] * v;
      }
    }
  for (int oc = 0; oc < 4; ++oc)
    z1e[(((size_t)b*4 + oc)*HH + y)*WW + x] = acc[oc];
}

// ---------------- weight swizzle into WMMA B-fragment layout ----------------
// FragB flat index = ((kc*128 + oc)*2 + half16)*16 + j ; K = kc*32 + (j<8 ? half16*8 + j : 16 + half16*8 + j-8)
__global__ void prep_weights(const float* __restrict__ Wxh, const float* __restrict__ Whh,
                             _Float16* __restrict__ FragB) {
  int tid = blockIdx.x * 256 + threadIdx.x;
  if (tid >= NCHUNK * 128 * 32) return;
  int kc  = tid / (128 * 32);
  int rem = tid % (128 * 32);
  int oc  = rem >> 5;
  int kic = ((rem & 15) < 8) ? (((rem >> 4) & 1)*8 + (rem & 15)) : (16 + ((rem >> 4) & 1)*8 + (rem & 15) - 8);  // K within chunk (table order for B layout)
  float val = 0.0f;
  if (kc < 25) {                  // h-part: one 5x5 tap, 32 channels
    int ky = kc / 5, kx = kc % 5;
    val = Whh[((oc*32 + kic)*5 + ky)*5 + kx];
  } else {                        // x-part: flattened (ic,tap), 125 valid
    int kp = (kc - 25) * 32 + kic;
    if (kp < 125) {
      int ic = kp / 25, tap = kp % 25;
      val = Wxh[((oc*5 + ic)*5 + tap/5)*5 + tap%5];
    }
  }
  FragB[tid] = (_Float16)val;
}

// ---------------- pack x-channels (shifted z2 + z1 embedding) into padded f16 buffer ----------------
__global__ void pack_x(const float* __restrict__ z2, const float* __restrict__ z1e,
                       _Float16* __restrict__ xh, int t) {
  int idx = blockIdx.x * 256 + threadIdx.x;
  if (idx >= BB * HH * WW) return;
  int x = idx % WW, y = (idx / WW) % HH, b = idx / (WW * HH);
  size_t base = (((size_t)b*PW + (y + 4))*PW + (x + 4))*CP;
  float sh = (t == 0) ? 0.0f : z2[(((size_t)b*SS + (t - 1))*HH + y)*WW + x];
  xh[base + 32] = (_Float16)sh;
#pragma unroll
  for (int c = 0; c < 4; ++c)
    xh[base + 33 + c] = (_Float16)z1e[(((size_t)b*4 + c)*HH + y)*WW + x];
}

// ---------------- fused ConvLSTM step: WMMA gates + pointwise + 1x1 out + LL ----------------
__launch_bounds__(256)
__global__ void lstm_step(const _Float16* __restrict__ xh_cur,
                          _Float16* __restrict__ xh_nxt,
                          float* __restrict__ cbuf,
                          const _Float16* __restrict__ FragB,
                          const float* __restrict__ bh,
                          const float* __restrict__ Wout,
                          const float* __restrict__ bout,
                          const float* __restrict__ z2,
                          float* __restrict__ out,
                          int t) {
  extern __shared__ __align__(16) unsigned char smem[];
  _Float16* sPatch = (_Float16*)smem;                         // 9 x 72 x 40 halves
  float*    sG     = (float*)(smem + SMEM_PATCH_BYTES);       // 64 pixels x 128 gates
  unsigned short* sXoff = (unsigned short*)(smem + SMEM_PATCH_BYTES + SMEM_G_BYTES);
  float*    sH     = (float*)smem;                            // reuse patch: 64 x 32 f32
  float*    sLL    = (float*)(smem + 64*EMB*4);               // 64 floats (still in patch region)

  const int tid = threadIdx.x;
  const int bid = blockIdx.x;
  const int b = bid >> 6;
  const int y = bid & 63;

  // ---- stage contiguous padded patch (rows y..y+8, all 72 cols, 40 ch) into LDS ----
#if HAVE_TDM
  if (tid < 32) {   // wave-uniform: wave 0 issues one TDM DMA for the whole patch
    unsigned long long ga =
        (unsigned long long)(uintptr_t)(xh_cur + ((size_t)(b*PW + y)) * PW * CP);
    unsigned int lds = (unsigned int)(uintptr_t)sPatch;
    v4u g0;
    g0.x = 1u;                                    // count=1 (valid user descriptor)
    g0.y = lds;                                   // lds_addr
    g0.z = (unsigned int)ga;                      // global_addr[31:0]
    g0.w = (unsigned int)((ga >> 32) & 0x01FFFFFFull) | (2u << 30); // addr[56:32] | type=2
    const unsigned int NH = (unsigned int)PATCH_HALVES;  // 25,920 elements of 2B
    v8i g1;
    g1[0] = (int)(1u << 16);                      // workgroup_mask=0, data_size=1 (2 bytes)
    g1[1] = (int)((NH & 0xFFFFu) << 16);          // tensor_dim0[15:0]
    g1[2] = (int)(((NH >> 16) & 0xFFFFu) | (1u << 16)); // tensor_dim0[31:16] | tensor_dim1=1
    g1[3] = (int)((NH & 0xFFFFu) << 16);          // tensor_dim1[31:16]=0 | tile_dim0
    g1[4] = 0;                                    // tile_dim1=0, tile_dim2=0 (1-D tile)
    g1[5] = (int)NH;                              // tensor_dim0_stride[31:0]
    g1[6] = 0;                                    // stride hi / tensor_dim1_stride lo
    g1[7] = 0;
    v4i gz = {0, 0, 0, 0};
#if TDM_SIX_ARG
    v8i gz8 = {0, 0, 0, 0, 0, 0, 0, 0};
    __builtin_amdgcn_tensor_load_to_lds(g0, g1, gz, gz, gz8, 0);
#else
    __builtin_amdgcn_tensor_load_to_lds(g0, g1, gz, gz, 0);
#endif
#if defined(__has_builtin)
#if __has_builtin(__builtin_amdgcn_s_wait_tensorcnt)
    __builtin_amdgcn_s_wait_tensorcnt(0);
#endif
#endif
  }
#else
  {
    const uint4* src = (const uint4*)(xh_cur + ((size_t)(b*PW + y)) * PW * CP);
    uint4* dst = (uint4*)sPatch;
    for (int i = tid; i < SMEM_PATCH_BYTES / 16; i += 256) dst[i] = src[i];
  }
#endif

  // ---- x-part gather offset table: kp -> patch offset (invalid -> zero pad channel 37) ----
  if (tid < 128) {
    int kp = tid;
    int ic = kp / 25, tap = kp - ic*25;
    unsigned short off = (unsigned short)(((2*(tap/5))*PW + 2*(tap%5))*CP + 32 + ic);
    sXoff[tid] = (kp < 125) ? off : (unsigned short)37;
  }
  __syncthreads();

  const int w  = tid >> 5;        // wave id -> N tile (16 gate channels)
  const int l  = tid & 31;
  const int lm = l & 15;
  const int lh = l >> 4;
  const int oc = w * 16 + lm;
  const int base8 = lh * 8;       // A-fragment K-base per lane half

  v8f acc0 = {}, acc1 = {}, acc2 = {}, acc3 = {};

  const _Float16* bbase = FragB + ((size_t)oc*2 + lh)*16;  // + kc*4096 per chunk

  // ---- h-part: 25 taps, K=32 channels each (b128 LDS reads, all aligned) ----
#pragma clang loop unroll(disable)
  for (int kc = 0; kc < 25; ++kc) {
    const _Float16* bp = bbase + (size_t)kc * 4096;
    __builtin_prefetch(bp + 4*4096, 0, 3);        // weight stream 4 chunks ahead (near cache)
    v16h bf = *(const v16h*)bp;
    int ky = kc / 5, kx = kc - ky*5;
    int abase = ((2*ky)*PW + 2*kx + lm)*CP + base8;
    {
      v8h lo = *(const v8h*)(sPatch + abase);
      v8h hi = *(const v8h*)(sPatch + abase + 16);
      v16h a;
#pragma unroll
      for (int e = 0; e < 8; ++e) { a[e] = lo[e]; a[8+e] = hi[e]; }
      acc0 = __builtin_amdgcn_wmma_f32_16x16x32_f16(false, a, false, bf, (short)0, acc0, false, false);
    }
    {
      v8h lo = *(const v8h*)(sPatch + abase + 640);
      v8h hi = *(const v8h*)(sPatch + abase + 656);
      v16h a;
#pragma unroll
      for (int e = 0; e < 8; ++e) { a[e] = lo[e]; a[8+e] = hi[e]; }
      acc1 = __builtin_amdgcn_wmma_f32_16x16x32_f16(false, a, false, bf, (short)0, acc1, false, false);
    }
    {
      v8h lo = *(const v8h*)(sPatch + abase + 1280);
      v8h hi = *(const v8h*)(sPatch + abase + 1296);
      v16h a;
#pragma unroll
      for (int e = 0; e < 8; ++e) { a[e] = lo[e]; a[8+e] = hi[e]; }
      acc2 = __builtin_amdgcn_wmma_f32_16x16x32_f16(false, a, false, bf, (short)0, acc2, false, false);
    }
    {
      v8h lo = *(const v8h*)(sPatch + abase + 1920);
      v8h hi = *(const v8h*)(sPatch + abase + 1936);
      v16h a;
#pragma unroll
      for (int e = 0; e < 8; ++e) { a[e] = lo[e]; a[8+e] = hi[e]; }
      acc3 = __builtin_amdgcn_wmma_f32_16x16x32_f16(false, a, false, bf, (short)0, acc3, false, false);
    }
  }

  // ---- x-part: 4 chunks over flattened (ic, tap); table-driven gathers, no per-lane division ----
#pragma unroll
  for (int kc = 25; kc < 29; ++kc) {
    v16h bf = *(const v16h*)(bbase + (size_t)kc * 4096);
    int tb[16];
#pragma unroll
    for (int e = 0; e < 16; ++e) {
      int kk = base8 + (e < 8 ? e : e + 8);
      tb[e] = sXoff[(kc - 25) * 32 + kk];
    }
#pragma unroll
    for (int mt = 0; mt < 4; ++mt) {
      int xcp = (mt*16 + lm) * CP;
      v16h a;
#pragma unroll
      for (int e = 0; e < 16; ++e) a[e] = sPatch[tb[e] + xcp];
      v8f c = (mt == 0) ? acc0 : (mt == 1) ? acc1 : (mt == 2) ? acc2 : acc3;
      c = __builtin_amdgcn_wmma_f32_16x16x32_f16(false, a, false, bf, (short)0, c, false, false);
      if (mt == 0) acc0 = c; else if (mt == 1) acc1 = c; else if (mt == 2) acc2 = c; else acc3 = c;
    }
  }

  // ---- spill gates (+bias) to LDS: D element (lane l, vgpr j) = pixel j+8*lh, channel oc ----
  float bias = bh[oc];
  v8f accs[4] = { acc0, acc1, acc2, acc3 };
#pragma unroll
  for (int mt = 0; mt < 4; ++mt)
#pragma unroll
    for (int j = 0; j < 8; ++j) {
      int pix = mt*16 + j + 8*lh;
      sG[pix*128 + oc] = accs[mt][j] + bias;
    }
  __syncthreads();

  // ---- pointwise LSTM update (f32 state), write h to next packed buffer + LDS ----
#pragma unroll
  for (int it = 0; it < 8; ++it) {
    int idx = tid + it*256;
    int pix = idx >> 5;
    int e   = idx & 31;
    float gi = sG[pix*128 + e];
    float gf = sG[pix*128 + 32 + e];
    float gg = sG[pix*128 + 64 + e];
    float go = sG[pix*128 + 96 + e];
    size_t cidx = (((size_t)(b*HH + y))*WW + pix)*EMB + e;
    float c0 = cbuf[cidx];
    float cn = sigm(gf)*c0 + sigm(gi)*tanhf(gg);
    float h  = sigm(go)*tanhf(cn);
    cbuf[cidx] = cn;
    xh_nxt[(((size_t)(b*PW) + (y + 4))*PW + (pix + 4))*CP + e] = (_Float16)h;
    sH[pix*EMB + e] = h;
  }
  __syncthreads();

  // ---- 1x1 output conv + log-likelihood per pixel ----
  if (tid < 64) {
    int pix = tid;
    float mean = bout[0], logs = bout[1];
#pragma unroll
    for (int e = 0; e < EMB; ++e) {
      float h = sH[pix*EMB + e];
      mean += Wout[e]       * h;
      logs += Wout[EMB + e] * h;
    }
    float zv = z2[(((size_t)b*SS + t)*HH + y)*WW + pix];
    float d  = zv - mean;
    sLL[pix] = -0.5f * (2.0f*logs + d*d*__expf(-2.0f*logs) + C_LOG2PI);
  }
  __syncthreads();
  if (tid == 0) {
    float s = 0.0f;
    for (int i = 0; i < 64; ++i) s += sLL[i];
    atomicAdd(out + b, s);
  }
}

extern "C" void kernel_launch(void* const* d_in, const int* in_sizes, int n_in,
                              void* d_out, int out_size, void* d_ws, size_t ws_size,
                              hipStream_t stream) {
  const float* z1   = (const float*)d_in[0];
  const float* z2   = (const float*)d_in[1];
  const float* W1   = (const float*)d_in[2];
  const float* b1   = (const float*)d_in[3];
  const float* W2   = (const float*)d_in[4];
  const float* b2   = (const float*)d_in[5];
  const float* Wxh  = (const float*)d_in[6];
  const float* Whh  = (const float*)d_in[7];
  const float* bh   = (const float*)d_in[8];
  const float* Wout = (const float*)d_in[9];
  const float* bout = (const float*)d_in[10];
  float* out = (float*)d_out;

  char* ws = (char*)d_ws;
  float*    out1  = (float*)(ws + OFF_OUT1);
  float*    z1e   = (float*)(ws + OFF_Z1E);
  _Float16* FragB = (_Float16*)(ws + OFF_FRAGB);
  float*    cbuf  = (float*)(ws + OFF_CBUF);
  _Float16* xh0   = (_Float16*)(ws + OFF_XH0);
  _Float16* xh1   = (_Float16*)(ws + OFF_XH1);

  hipMemsetAsync(out, 0, 32 * sizeof(float), stream);   // ll accumulators
  hipMemsetAsync(cbuf, 0, SZ_CBUF, stream);             // c0 = 0
  hipMemsetAsync(xh0, 0, SZ_XH, stream);                // h0 = 0, zero halo + pad channels
  hipMemsetAsync(xh1, 0, SZ_XH, stream);

  prep_weights<<<(NCHUNK*128*32 + 255)/256, 256, 0, stream>>>(Wxh, Whh, FragB);
  conv_embed1<<<(BB*HH*WW + 255)/256, 256, 0, stream>>>(z1, W1, b1, out1);
  conv_embed2<<<(BB*HH*WW + 255)/256, 256, 0, stream>>>(out1, W2, b2, z1e);

  _Float16* bufs[2] = { xh0, xh1 };
  for (int t = 0; t < SS; ++t) {
    _Float16* cur = bufs[t & 1];
    _Float16* nxt = bufs[(t + 1) & 1];
    pack_x<<<(BB*HH*WW + 255)/256, 256, 0, stream>>>(z2, z1e, cur, t);
    lstm_step<<<BB*HH, 256, SMEM_BYTES, stream>>>(cur, nxt, cbuf, FragB, bh,
                                                  Wout, bout, z2, out, t);
  }
}
